// ViTForImageRotation_54056458387743
// MI455X (gfx1250) — matrix operand order repaired and checked by reading, not compiled, not run on hardware
//
#include <hip/hip_runtime.h>
#include <hip/hip_bf16.h>

typedef __attribute__((ext_vector_type(16))) _Float16 v16h;
typedef __attribute__((ext_vector_type(8)))  _Float16 v8h;
typedef __attribute__((ext_vector_type(8)))  float    v8f;

#define DEV __device__ __forceinline__

#define BATCH  64
#define SEQ    197
#define HIDD   768
#define NHEAD  12
#define DHEAD  64
#define DFFD   3072
#define OUTDD  1000
#define NPATCH 196
#define LAYERS 4
#define MTOK   (BATCH*SEQ)     /* 12608, multiple of 64 */
#define MPAT   (BATCH*NPATCH)  /* 12544, multiple of 64 */
#define SPAD   224
#define VFRAG  (7*4*32*16)     /* 14336 halves per (b,h) of packed V */

// ---------------- WMMA helpers (CDNA5 16x16x32 f16 -> f32) ----------------
DEV v8f wmma_f16(v16h a, v16h b, v8f c) {
  return __builtin_amdgcn_wmma_f32_16x16x32_f16(false, a, false, b, (short)0, c, false, false);
}

DEV v16h cat8(v8h lo, v8h hi) {
  return __builtin_shufflevector(lo, hi, 0,1,2,3,4,5,6,7,8,9,10,11,12,13,14,15);
}

// A fragment: 16x32 tile of row-major f16 matrix, base at (row0,k0), ld = lda (halves).
// lanes 0-15 (M=0..15): K 0..15 ; lanes 16-31: K 16..31. 16B-aligned runs.
DEV v16h load_a_rm(const _Float16* A, int lda) {
  int lane = threadIdx.x & 31;
  int row  = lane & 15;
  int kb   = (lane & 16) ? 16 : 0;
  const _Float16* p = A + (size_t)row * lda + kb;
  return cat8(*(const v8h*)p, *(const v8h*)(p + 8));
}

DEV v16h load_a_rm_g(const _Float16* A, int lda, int rows_rem) {
  int lane = threadIdx.x & 31;
  int row  = lane & 15;
  int kb   = (lane & 16) ? 16 : 0;
  v16h a;
  if (row < rows_rem) {
    const _Float16* p = A + (size_t)row * lda + kb;
    a = cat8(*(const v8h*)p, *(const v8h*)(p + 8));
  } else {
#pragma unroll
    for (int i = 0; i < 16; ++i) a[i] = (_Float16)0.f;
  }
  return a;
}

// B fragment from K-contiguous storage: lane reads 16 contiguous halves at
// base[(n-local column) selected by caller]. p must be 16B aligned.
DEV v16h load_b_kcontig(const _Float16* p) {
  return cat8(*(const v8h*)p, *(const v8h*)(p + 8));
}

// B fragment where B = transpose of row-major SxK matrix: B[kk][n]=Ksrc[n][kk].
// One aligned 32B read per lane.
DEV v16h load_bT_g(const _Float16* Ksrc, int ldk, int cols_rem) {
  int lane = threadIdx.x & 31;
  int col  = lane & 15;
  int kb   = (lane & 16) ? 16 : 0;
  v16h b;
  if (col < cols_rem) {
    b = *(const v16h*)(Ksrc + (size_t)col * ldk + kb);
  } else {
#pragma unroll
    for (int i = 0; i < 16; ++i) b[i] = (_Float16)0.f;
  }
  return b;
}

DEV void stv(float* p, float v)     { *p = v; }
DEV void stv(_Float16* p, float v)  { *p = (_Float16)v; }

// ---------------- elementwise / prep kernels ----------------
__global__ __launch_bounds__(256) void k_f32_to_f16(const float* __restrict__ s,
                                                    _Float16* __restrict__ d, int n) {
  int i = blockIdx.x * 256 + threadIdx.x;
  if (i < n) d[i] = (_Float16)s[i];
}

// Wq [L,H,e,d] f32 -> per-head WMMA B-fragment layout:
// dst[(l*H+h)*4096 + ((ks*4+nt)*32 + lane)*16 + ii], ks=d>>5, r=d&31,
// lane=(r&16)|(e&15), ii=r&15, nt=e>>4.  => B frag is one aligned v16h load.
__global__ __launch_bounds__(256) void k_convert_headw(const float* __restrict__ s,
                                                       _Float16* __restrict__ d, int n) {
  int i = blockIdx.x * 256 + threadIdx.x;
  if (i >= n) return;
  int dd = i & 63, e = (i >> 6) & 63, h = (i >> 12) % NHEAD, l = i / (64 * 64 * NHEAD);
  int ks = dd >> 5, r = dd & 31;
  int lane = (r & 16) | (e & 15);
  int ii = r & 15, nt = e >> 4;
  d[(size_t)(l * NHEAD + h) * 4096 + ((ks * 4 + nt) * 32 + lane) * 16 + ii] = (_Float16)s[i];
}

// pack V [B*H, S, 64] f16 -> fragment layout [B*H][ks(7)][nt(4)][lane(32)][i(16)], zero pad
__global__ __launch_bounds__(256) void k_pack_v(const _Float16* __restrict__ v,
                                                _Float16* __restrict__ vf) {
  int idx = blockIdx.x * 256 + threadIdx.x;
  if (idx >= BATCH * NHEAD * VFRAG) return;
  int i = idx & 15, lane = (idx >> 4) & 31, nt = (idx >> 9) & 3;
  int ks = (idx >> 11) % 7, bh = idx / VFRAG;
  int row = ks * 32 + (lane & 16) + i;
  int col = nt * 16 + (lane & 15);
  vf[idx] = (row < SEQ) ? v[((size_t)bh * SEQ + row) * DHEAD + col] : (_Float16)0.f;
}

// images [B,3,224,224] f32 -> patches [B*196, 768] f16  (c,py,px order)
__global__ __launch_bounds__(256) void k_patchify(const float* __restrict__ img,
                                                  _Float16* __restrict__ p) {
  int i = blockIdx.x * 256 + threadIdx.x;
  if (i >= MPAT * HIDD) return;
  int col = i % HIDD, pp = (i / HIDD) % NPATCH, b = i / (HIDD * NPATCH);
  int c = col >> 8, rem = col & 255, py = rem >> 4, px = rem & 15;
  int pi = pp / 14, pj = pp % 14;
  p[i] = (_Float16)img[(((size_t)(b * 3 + c) * 224) + (pi * 16 + py)) * 224 + (pj * 16 + px)];
}

// out[b,s,:] = (s==0 ? cls : tokens[b,s-1]) + pos(s,:)
__global__ __launch_bounds__(256) void k_assemble(const float* __restrict__ tok,
                                                  const float* __restrict__ cls,
                                                  float* __restrict__ out) {
  int i = blockIdx.x * 256 + threadIdx.x;
  if (i >= MTOK * HIDD) return;
  int d = i % HIDD, s = (i / HIDD) % SEQ, b = i / (HIDD * SEQ);
  float v = (s == 0) ? cls[d] : tok[((size_t)(b * NPATCH + s - 1)) * HIDD + d];
  float ang = (float)s * __powf(10000.f, -(float)((d >> 1) * 2) / (float)HIDD);
  v += ((d & 1) == 0) ? __sinf(ang) : __cosf(ang);
  out[i] = v;
}

// LayerNorm row kernel: x f32 [M,768] -> y f16 [M,768]
__global__ __launch_bounds__(256) void k_ln(const float* __restrict__ x,
                                            const float* __restrict__ g,
                                            const float* __restrict__ bt,
                                            _Float16* __restrict__ y) {
  __shared__ float s1[256], s2[256];
  int row = blockIdx.x;
  const float* xr = x + (size_t)row * HIDD;
  float a = 0.f, b = 0.f;
  for (int i = threadIdx.x; i < HIDD; i += 256) { float v = xr[i]; a += v; b += v * v; }
  s1[threadIdx.x] = a; s2[threadIdx.x] = b; __syncthreads();
  for (int st = 128; st > 0; st >>= 1) {
    if (threadIdx.x < st) { s1[threadIdx.x] += s1[threadIdx.x + st]; s2[threadIdx.x] += s2[threadIdx.x + st]; }
    __syncthreads();
  }
  float m = s1[0] / (float)HIDD;
  float var = s2[0] / (float)HIDD - m * m;
  float inv = rsqrtf(var + 1e-5f);
  for (int i = threadIdx.x; i < HIDD; i += 256)
    y[(size_t)row * HIDD + i] = (_Float16)((xr[i] - m) * inv * g[i] + bt[i]);
}

__global__ __launch_bounds__(256) void k_take_cls(const float* __restrict__ out32,
                                                  _Float16* __restrict__ c16) {
  int i = blockIdx.x * 256 + threadIdx.x;
  if (i >= BATCH * HIDD) return;
  int d = i % HIDD, b = i / HIDD;
  c16[i] = (_Float16)out32[((size_t)b * SEQ) * HIDD + d];
}

__global__ __launch_bounds__(256) void k_softmax_row(const float* __restrict__ lg,
                                                     float* __restrict__ out, int N) {
  __shared__ float s1[256];
  int row = blockIdx.x;
  const float* xr = lg + (size_t)row * N;
  float mx = -1e30f;
  for (int i = threadIdx.x; i < N; i += 256) mx = fmaxf(mx, xr[i]);
  s1[threadIdx.x] = mx; __syncthreads();
  for (int st = 128; st > 0; st >>= 1) {
    if (threadIdx.x < st) s1[threadIdx.x] = fmaxf(s1[threadIdx.x], s1[threadIdx.x + st]);
    __syncthreads();
  }
  float rmax = s1[0]; __syncthreads();
  float sm = 0.f;
  for (int i = threadIdx.x; i < N; i += 256) sm += __expf(xr[i] - rmax);
  s1[threadIdx.x] = sm; __syncthreads();
  for (int st = 128; st > 0; st >>= 1) {
    if (threadIdx.x < st) s1[threadIdx.x] += s1[threadIdx.x + st];
    __syncthreads();
  }
  float inv = 1.f / s1[0];
  for (int i = threadIdx.x; i < N; i += 256)
    out[(size_t)row * N + i] = __expf(xr[i] - rmax) * inv;
}

// ---------------- generic WMMA GEMM: out = op(A@B + bias) (+resid) ----------------
// A f16 [M,K] rm, B f16 [K,N] rm. M%64==0, K%32==0. N guarded. 256 thr, 8 waves,
// block tile 64x128, wave tile 16x64. A tile filled with async-to-LDS b128 DMA,
// B tile stored transposed in LDS so fragments are K-contiguous b128 reads.
template <bool HASBIAS, bool DOGELU, bool DORESID, typename TOUT>
__global__ __launch_bounds__(256) void k_gemm(const _Float16* __restrict__ A,
                                              const _Float16* __restrict__ B,
                                              const float* __restrict__ bias,
                                              const float* __restrict__ resid,
                                              TOUT* __restrict__ out,
                                              int M, int N, int K) {
  __shared__ alignas(16) _Float16 As[64][40];
  __shared__ alignas(16) _Float16 Bt[128][40];   // Bt[col][k]
  int tid = threadIdx.x;
  int bm = blockIdx.x * 64;
  int bn = blockIdx.y * 128;
  int w = tid >> 5;
  int wm = (w & 3) * 16;
  int wn = (w >> 2) * 64;
  v8f acc[4] = {};

  int ar = tid >> 2, ac = (tid & 3) * 8;          // A fill: 8 halves / thread
  unsigned ldsA = (unsigned)(size_t)(void*)&As[ar][ac];
  const bool fullN = (bn + 128 <= N);

  for (int k0 = 0; k0 < K; k0 += 32) {
    // --- async DMA the A tile (64x32) into LDS: one b128 per thread ---
    {
      unsigned long long ga =
          (unsigned long long)(const void*)(A + (size_t)(bm + ar) * K + k0 + ac);
      asm volatile("global_load_async_to_lds_b128 %0, %1, off"
                   :: "v"(ldsA), "v"(ga) : "memory");
    }
    // --- fill B tile transposed (32x128 -> Bt[col][k]) ---
#pragma unroll
    for (int sgi = 0; sgi < 2; ++sgi) {
      int sg = tid + sgi * 256;
      int br = sg >> 4, bc = (sg & 15) * 8;
      if (fullN) {
        v8h bv = *(const v8h*)(B + (size_t)(k0 + br) * N + bn + bc);
        if (k0 + 32 < K)
          __builtin_prefetch(B + (size_t)(k0 + 32 + br) * N + bn + bc, 0, 0);
#pragma unroll
        for (int i = 0; i < 8; ++i) Bt[bc + i][br] = bv[i];
      } else {
#pragma unroll
        for (int i = 0; i < 8; ++i) {
          int col = bn + bc + i;
          Bt[bc + i][br] = (col < N) ? B[(size_t)(k0 + br) * N + col] : (_Float16)0.f;
        }
      }
    }
    asm volatile("s_wait_asynccnt 0x0" ::: "memory");
    __syncthreads();

    {
      v16h af = load_a_rm(&As[wm][0], 40);
      int lane = tid & 31;
      int col = lane & 15;
      int kb  = (lane & 16) ? 16 : 0;
#pragma unroll
      for (int j = 0; j < 4; ++j)
        acc[j] = wmma_f16(af, load_b_kcontig(&Bt[wn + 16 * j + col][kb]), acc[j]);
    }
    __syncthreads();
  }

  int lane = tid & 31;
  int cl = lane & 15;
  int rb = (lane & 16) ? 8 : 0;
#pragma unroll
  for (int j = 0; j < 4; ++j) {
    int col = bn + wn + j * 16 + cl;
    if (col < N) {
#pragma unroll
      for (int r = 0; r < 8; ++r) {
        int row = bm + wm + rb + r;
        float v = acc[j][r];
        if (HASBIAS) v += bias[col];
        if (DOGELU)  v = 0.5f * v * (1.f + erff(v * 0.70710678118f));
        if (DORESID) v += resid[(size_t)row * N + col];
        stv(&out[(size_t)row * N + col], v);
      }
    }
  }
}

// ---------------- QKV: per-head 64x64 GEMM with bias (B pre-packed frags) ----------------
// grid.x = MTOK/32, grid.y = 3*NHEAD. qkv layout [3][B][H][S][64] f16.
__global__ __launch_bounds__(256) void k_qkv(const _Float16* __restrict__ xln,
                                             const _Float16* __restrict__ wqf,
                                             const _Float16* __restrict__ wkf,
                                             const _Float16* __restrict__ wvf,
                                             const float* __restrict__ bq,
                                             const float* __restrict__ bk,
                                             const float* __restrict__ bv,
                                             _Float16* __restrict__ qkv) {
  int which = blockIdx.y / NHEAD;
  int h     = blockIdx.y % NHEAD;
  int row0  = blockIdx.x * 32;
  const _Float16* W   = (which == 0) ? wqf : (which == 1) ? wkf : wvf;
  const float*    bia = (which == 0) ? bq  : (which == 1) ? bk  : bv;
  W += (size_t)h * 4096;
  _Float16* dst = qkv + (size_t)which * BATCH * NHEAD * SEQ * DHEAD;

  int w = threadIdx.x >> 5;
  int wm = (w & 1) * 16;
  int wnt = (w >> 1);                    // n-tile 0..3
  int lane = threadIdx.x & 31;
  v8f acc = {};
#pragma unroll
  for (int ks = 0; ks < 2; ++ks) {
    v16h af = load_a_rm(xln + (size_t)(row0 + wm) * HIDD + h * DHEAD + ks * 32, HIDD);
    v16h bf = *(const v16h*)(W + ((ks * 4 + wnt) * 32 + lane) * 16);
    acc = wmma_f16(af, bf, acc);
  }
  int cl = lane & 15;
  int rb = (lane & 16) ? 8 : 0;
  int e  = wnt * 16 + cl;
  float bb = bia[h * DHEAD + e];
#pragma unroll
  for (int r = 0; r < 8; ++r) {
    int rr = row0 + wm + rb + r;
    int b = rr / SEQ, s = rr % SEQ;
    dst[(((size_t)(b * NHEAD + h) * SEQ) + s) * DHEAD + e] = (_Float16)(acc[r] + bb);
  }
}

// ---------------- attention scores + softmax: att[b,h] = softmax(q k^T / 8) ----------------
// grid.x = B*H, grid.y = 7 (32 rows each). att f16 [B*H, 224, 224], zero-padded.
__global__ __launch_bounds__(256) void k_scores(const _Float16* __restrict__ q,
                                                const _Float16* __restrict__ k,
                                                _Float16* __restrict__ att) {
  __shared__ float sc[32][SPAD];
  __shared__ float red[32][8];
  int bh = blockIdx.x;
  int row0 = blockIdx.y * 32;
  const _Float16* qp = q + (size_t)bh * SEQ * DHEAD;
  const _Float16* kp = k + (size_t)bh * SEQ * DHEAD;
  _Float16* ap = att + (size_t)bh * SPAD * SPAD;
  int tid = threadIdx.x, w = tid >> 5;

  for (int t = w; t < 28; t += 8) {
    int tm = t & 1, tn = t >> 1;           // tn in [0,14)
    v8f acc = {};
#pragma unroll
    for (int ks = 0; ks < 2; ++ks) {
      v16h af = load_a_rm_g(qp + (size_t)(row0 + tm * 16) * DHEAD + ks * 32, DHEAD,
                            SEQ - (row0 + tm * 16));
      v16h bf = load_bT_g(kp + (size_t)(tn * 16) * DHEAD + ks * 32, DHEAD, SEQ - tn * 16);
      acc = wmma_f16(af, bf, acc);
    }
    int lane = tid & 31, cl = lane & 15, rb = (lane & 16) ? 8 : 0;
#pragma unroll
    for (int r = 0; r < 8; ++r)
      sc[tm * 16 + rb + r][tn * 16 + cl] = acc[r] * 0.125f;
  }
  __syncthreads();

  // softmax over 197 valid cols; 8 threads per row
  int r = tid >> 3, j = tid & 7;
  float mx = -1e30f;
  for (int c = j; c < SEQ; c += 8) mx = fmaxf(mx, sc[r][c]);
  red[r][j] = mx; __syncthreads();
  float rmax = red[r][0];
#pragma unroll
  for (int jj = 1; jj < 8; ++jj) rmax = fmaxf(rmax, red[r][jj]);
  __syncthreads();
  float sm = 0.f;
  for (int c = j; c < SEQ; c += 8) sm += __expf(sc[r][c] - rmax);
  red[r][j] = sm; __syncthreads();
  float rsum = 0.f;
#pragma unroll
  for (int jj = 0; jj < 8; ++jj) rsum += red[r][jj];
  float inv = 1.f / rsum;
  int srow = row0 + r;
  for (int c = j; c < SPAD; c += 8) {
    float v = (srow < SEQ && c < SEQ) ? __expf(sc[r][c] - rmax) * inv : 0.f;
    ap[(size_t)srow * SPAD + c] = (_Float16)v;
  }
}

// ---------------- o = att @ v (packed V frags), residual add into out32 ----------------
// grid.x = B*H, grid.y = 7 (32 rows).
__global__ __launch_bounds__(256) void k_attv(const _Float16* __restrict__ att,
                                              const _Float16* __restrict__ vf,
                                              float* __restrict__ out32) {
  int bh = blockIdx.x;
  int row0 = blockIdx.y * 32;
  int bb = bh / NHEAD, hh = bh % NHEAD;
  const _Float16* ap  = att + (size_t)bh * SPAD * SPAD;
  const _Float16* vfp = vf + (size_t)bh * VFRAG;
  int w = threadIdx.x >> 5;
  int wm = (w & 1) * 16;
  int wnt = (w >> 1);
  int lane = threadIdx.x & 31;
  v8f acc = {};
#pragma unroll
  for (int ks = 0; ks < 7; ++ks) {
    v16h af = load_a_rm(ap + (size_t)(row0 + wm) * SPAD + ks * 32, SPAD);
    v16h bf = *(const v16h*)(vfp + ((ks * 4 + wnt) * 32 + lane) * 16);
    acc = wmma_f16(af, bf, acc);
  }
  int cl = lane & 15, rb = (lane & 16) ? 8 : 0;
#pragma unroll
  for (int r = 0; r < 8; ++r) {
    int s = row0 + wm + rb + r;
    if (s < SEQ) {
      size_t idx = ((size_t)(bb * SEQ + s)) * HIDD + hh * DHEAD + wnt * 16 + cl;
      out32[idx] += acc[r];
    }
  }
}

// ---------------- host launch ----------------
extern "C" void kernel_launch(void* const* d_in, const int* in_sizes, int n_in,
                              void* d_out, int out_size, void* d_ws, size_t ws_size,
                              hipStream_t stream) {
  const float* images = (const float*)d_in[0];
  const float* Wp    = (const float*)d_in[1];
  const float* bp    = (const float*)d_in[2];
  const float* cls   = (const float*)d_in[3];
  const float* ln1_g = (const float*)d_in[4];
  const float* ln1_b = (const float*)d_in[5];
  const float* Wq    = (const float*)d_in[6];
  const float* bq    = (const float*)d_in[7];
  const float* Wk    = (const float*)d_in[8];
  const float* bk    = (const float*)d_in[9];
  const float* Wv    = (const float*)d_in[10];
  const float* bv    = (const float*)d_in[11];
  const float* ln2_g = (const float*)d_in[12];
  const float* ln2_b = (const float*)d_in[13];
  const float* W1    = (const float*)d_in[14];
  const float* b1    = (const float*)d_in[15];
  const float* W2    = (const float*)d_in[16];
  const float* b2    = (const float*)d_in[17];
  const float* Wo    = (const float*)d_in[18];
  const float* bo    = (const float*)d_in[19];
  float* dout = (float*)d_out;

  char* wsp = (char*)d_ws;
  auto alloc = [&](size_t bytes) -> void* {
    void* p = (void*)wsp;
    wsp += (bytes + 255) & ~(size_t)255;
    return p;
  };

  _Float16* wp16   = (_Float16*)alloc((size_t)HIDD * HIDD * 2);
  _Float16* wq16f  = (_Float16*)alloc((size_t)LAYERS * NHEAD * 64 * 64 * 2);
  _Float16* wk16f  = (_Float16*)alloc((size_t)LAYERS * NHEAD * 64 * 64 * 2);
  _Float16* wv16f  = (_Float16*)alloc((size_t)LAYERS * NHEAD * 64 * 64 * 2);
  _Float16* w116   = (_Float16*)alloc((size_t)LAYERS * HIDD * DFFD * 2);
  _Float16* w216   = (_Float16*)alloc((size_t)LAYERS * DFFD * HIDD * 2);
  _Float16* wo16   = (_Float16*)alloc((size_t)HIDD * OUTDD * 2);
  _Float16* pat16  = (_Float16*)alloc((size_t)MPAT * HIDD * 2);
  float*    tok32  = (float*)   alloc((size_t)MPAT * HIDD * 4);
  float*    out32  = (float*)   alloc((size_t)MTOK * HIDD * 4);
  _Float16* xln16  = (_Float16*)alloc((size_t)MTOK * HIDD * 2);
  _Float16* qkv16  = (_Float16*)alloc((size_t)3 * BATCH * NHEAD * SEQ * DHEAD * 2);
  _Float16* vfrag  = (_Float16*)alloc((size_t)BATCH * NHEAD * VFRAG * 2);
  _Float16* att16  = (_Float16*)alloc((size_t)BATCH * NHEAD * SPAD * SPAD * 2);
  _Float16* h116   = (_Float16*)alloc((size_t)MTOK * DFFD * 2);
  _Float16* cls16  = (_Float16*)alloc((size_t)BATCH * HIDD * 2);
  float*    log32  = (float*)   alloc((size_t)BATCH * OUTDD * 4);
  (void)ws_size; (void)n_in; (void)in_sizes; (void)out_size;

  auto g1 = [](int n) { return dim3((n + 255) / 256); };

  // weight conversions
  k_f32_to_f16<<<g1(HIDD * HIDD), 256, 0, stream>>>(Wp, wp16, HIDD * HIDD);
  k_f32_to_f16<<<g1(LAYERS * HIDD * DFFD), 256, 0, stream>>>(W1, w116, LAYERS * HIDD * DFFD);
  k_f32_to_f16<<<g1(LAYERS * DFFD * HIDD), 256, 0, stream>>>(W2, w216, LAYERS * DFFD * HIDD);
  k_f32_to_f16<<<g1(HIDD * OUTDD), 256, 0, stream>>>(Wo, wo16, HIDD * OUTDD);
  int nhw = LAYERS * NHEAD * 64 * 64;
  k_convert_headw<<<g1(nhw), 256, 0, stream>>>(Wq, wq16f, nhw);
  k_convert_headw<<<g1(nhw), 256, 0, stream>>>(Wk, wk16f, nhw);
  k_convert_headw<<<g1(nhw), 256, 0, stream>>>(Wv, wv16f, nhw);

  // patch embedding
  k_patchify<<<g1(MPAT * HIDD), 256, 0, stream>>>(images, pat16);
  k_gemm<true, false, false, float><<<dim3(MPAT / 64, HIDD / 128), 256, 0, stream>>>(
      pat16, wp16, bp, nullptr, tok32, MPAT, HIDD, HIDD);
  k_assemble<<<g1(MTOK * HIDD), 256, 0, stream>>>(tok32, cls, out32);

  _Float16* qb = qkv16;
  _Float16* kb = qkv16 + (size_t)BATCH * NHEAD * SEQ * DHEAD;
  _Float16* vb = qkv16 + (size_t)2 * BATCH * NHEAD * SEQ * DHEAD;

  for (int l = 0; l < LAYERS; ++l) {
    k_ln<<<dim3(MTOK), 256, 0, stream>>>(out32, ln1_g + l * HIDD, ln1_b + l * HIDD, xln16);
    k_qkv<<<dim3(MTOK / 32, 3 * NHEAD), 256, 0, stream>>>(
        xln16,
        wq16f + (size_t)l * NHEAD * 4096,
        wk16f + (size_t)l * NHEAD * 4096,
        wv16f + (size_t)l * NHEAD * 4096,
        bq + (size_t)l * NHEAD * DHEAD, bk + (size_t)l * NHEAD * DHEAD,
        bv + (size_t)l * NHEAD * DHEAD, qkv16);
    k_pack_v<<<g1(BATCH * NHEAD * VFRAG), 256, 0, stream>>>(vb, vfrag);
    k_scores<<<dim3(BATCH * NHEAD, 7), 256, 0, stream>>>(qb, kb, att16);
    k_attv<<<dim3(BATCH * NHEAD, 7), 256, 0, stream>>>(att16, vfrag, out32);
    k_ln<<<dim3(MTOK), 256, 0, stream>>>(out32, ln2_g + l * HIDD, ln2_b + l * HIDD, xln16);
    k_gemm<true, true, false, _Float16><<<dim3(MTOK / 64, DFFD / 128), 256, 0, stream>>>(
        xln16, w116 + (size_t)l * HIDD * DFFD, b1 + (size_t)l * DFFD, nullptr, h116,
        MTOK, DFFD, HIDD);
    k_gemm<true, false, true, float><<<dim3(MTOK / 64, HIDD / 128), 256, 0, stream>>>(
        h116, w216 + (size_t)l * DFFD * HIDD, b2 + (size_t)l * HIDD, out32, out32,
        MTOK, HIDD, DFFD);
  }

  k_take_cls<<<g1(BATCH * HIDD), 256, 0, stream>>>(out32, cls16);
  k_gemm<true, false, false, float><<<dim3(BATCH / 64, (OUTDD + 127) / 128), 256, 0, stream>>>(
      cls16, wo16, bo, nullptr, log32, BATCH, OUTDD, HIDD);
  k_softmax_row<<<dim3(BATCH), 256, 0, stream>>>(log32, dout, OUTDD);
}
